// Qwen3Attention_19292993094381
// MI455X (gfx1250) — matrix operand order repaired and checked by reading, not compiled, not run on hardware
//
#include <hip/hip_runtime.h>
#include <stdint.h>

#define T_SEQ    2048
#define HIDDEN   2048
#define NUM_HEADS 16
#define NUM_KV    8
#define HEAD_DIM  128
#define QKV_OUT  ((NUM_HEADS + 2 * NUM_KV) * HEAD_DIM)  // 4096

typedef __attribute__((ext_vector_type(16))) __bf16 bf16x16;
typedef __attribute__((ext_vector_type(8)))  float  f32x8;

static __device__ __forceinline__ unsigned short f2bf(float f) {
  unsigned int u = __float_as_uint(f);
  unsigned int r = u + 0x7FFFu + ((u >> 16) & 1u);  // round-to-nearest-even
  return (unsigned short)(r >> 16);
}

union FragU { uint4 u[2]; bf16x16 v; };
static __device__ __forceinline__ bf16x16 mkfrag(uint4 lo, uint4 hi) {
  FragU f; f.u[0] = lo; f.u[1] = hi; return f.v;
}
static __device__ __forceinline__ f32x8 zero8() {
  f32x8 z = {0.f, 0.f, 0.f, 0.f, 0.f, 0.f, 0.f, 0.f};
  return z;
}

// Async copy 16B global -> LDS (CDNA5 GLOBAL_LOAD_ASYNC_TO_LDS_B128, ASYNCcnt)
static __device__ __forceinline__ void async_copy_b128(unsigned lds_off,
                                                       const void* gaddr) {
  asm volatile("global_load_async_to_lds_b128 %0, %1, off"
               :: "v"(lds_off), "v"(gaddr) : "memory");
}
static __device__ __forceinline__ void wait_async0() {
  asm volatile("s_wait_asynccnt 0x0" ::: "memory");
}

// ---------------------------------------------------------------------------
// 1) fp32 -> bf16 conversion (vectorized: float4 in, 4 x bf16 out)
// ---------------------------------------------------------------------------
__global__ void f32_to_bf16_kernel(const float* __restrict__ src,
                                   unsigned short* __restrict__ dst, int n4) {
  int i = blockIdx.x * blockDim.x + threadIdx.x;
  if (i < n4) {
    float4 v = ((const float4*)src)[i];
    union { unsigned short s[4]; uint2 u; } o;
    o.s[0] = f2bf(v.x); o.s[1] = f2bf(v.y);
    o.s[2] = f2bf(v.z); o.s[3] = f2bf(v.w);
    ((uint2*)dst)[i] = o.u;
  }
}

// ---------------------------------------------------------------------------
// 2) bf16 GEMM: C[M,N] (f32) = A[M,K] x B[K,N], row-major, WMMA 16x16x32.
//    256 thr = 8 waves; macro tile 128(M) x 64(N); wave tile 32x32; K-step 64.
//    Double-buffered LDS; A tile streamed with global_load_async_to_lds_b128.
// ---------------------------------------------------------------------------
#define KSTEP 64

__global__ __launch_bounds__(256) void gemm_bf16_kernel(
    const unsigned short* __restrict__ A, const unsigned short* __restrict__ B,
    float* __restrict__ C, int M, int N, int K) {
  __shared__ unsigned short As[2][128][72];  // M x K, K-contiguous, padded
  __shared__ unsigned short Bs[2][64][72];   // N x K (transposed), K-contiguous

  const int tid = threadIdx.x;
  const int lane = tid & 31;
  const int wid = tid >> 5;
  const int waveM = wid & 3;   // 4 waves along M
  const int waveN = wid >> 2;  // 2 waves along N
  const int m0 = blockIdx.y * 128;
  const int n0 = blockIdx.x * 64;
  const int sel = lane >> 4;   // lane half (0/1)
  const int l16 = lane & 15;

  // staging responsibilities
  const int arow = tid >> 1, acs = (tid & 1) * 32;  // A: 128 rows x 64 k
  const int brow = tid >> 2, bnc = (tid & 3) * 16;  // B: 64 k-rows x 64 n

  f32x8 acc[2][2];
  acc[0][0] = zero8(); acc[0][1] = zero8();
  acc[1][0] = zero8(); acc[1][1] = zero8();

  auto issue_tile = [&](int k0, int buf) {
    // A tile: 4 x 16B async copies per thread (no transpose needed)
    const unsigned short* ga = A + (size_t)(m0 + arow) * K + k0 + acs;
    unsigned lbase = (unsigned)(size_t)&As[buf][arow][acs];
#pragma unroll
    for (int i = 0; i < 4; ++i)
      async_copy_b128(lbase + 16u * i, ga + 8 * i);
    // B tile: 16 contiguous n-values at k-row `brow`, transposed into LDS
    const unsigned short* gb = B + (size_t)(k0 + brow) * N + n0 + bnc;
    uint4 b0 = *(const uint4*)gb;
    uint4 b1 = *(const uint4*)(gb + 8);
    union { uint4 u; unsigned short s[8]; } u0, u1;
    u0.u = b0; u1.u = b1;
#pragma unroll
    for (int i = 0; i < 8; ++i) Bs[buf][bnc + i][brow] = u0.s[i];
#pragma unroll
    for (int i = 0; i < 8; ++i) Bs[buf][bnc + 8 + i][brow] = u1.s[i];
  };

  issue_tile(0, 0);
  wait_async0();
  __syncthreads();

  const int nk = K / KSTEP;
  for (int kt = 0; kt < nk; ++kt) {
    const int buf = kt & 1;
    if (kt + 1 < nk) issue_tile((kt + 1) * KSTEP, buf ^ 1);  // prefetch next

#pragma unroll
    for (int kc = 0; kc < 2; ++kc) {
      bf16x16 af[2], bfm[2];
#pragma unroll
      for (int i = 0; i < 2; ++i) {  // A: row per lane, K split by lane half
        const unsigned short* p = &As[buf][waveM * 32 + i * 16 + l16][kc * 32];
        uint4 lo = *(const uint4*)(p + sel * 8);
        uint4 hi = *(const uint4*)(p + 16 + sel * 8);
        af[i] = mkfrag(lo, hi);
      }
#pragma unroll
      for (int j = 0; j < 2; ++j) {  // B: col per lane, K 8*sel..8*sel+7 / 16+8*sel..23+8*sel
        const unsigned short* p = &Bs[buf][waveN * 32 + j * 16 + l16][kc * 32];
        uint4 lo = *(const uint4*)(p + sel * 8);
        uint4 hi = *(const uint4*)(p + 16 + sel * 8);
        bfm[j] = mkfrag(lo, hi);
      }
#pragma unroll
      for (int i = 0; i < 2; ++i)
#pragma unroll
        for (int j = 0; j < 2; ++j)
          acc[i][j] = __builtin_amdgcn_wmma_f32_16x16x32_bf16(
              false, af[i], false, bfm[j], (short)0, acc[i][j], false, false);
    }

    if (kt + 1 < nk) {
      wait_async0();    // next A tile resident in LDS
      __syncthreads();  // + release of this wave's B transpose stores
    }
  }

  // epilogue: base pointer per tile, step by N per row
#pragma unroll
  for (int i = 0; i < 2; ++i)
#pragma unroll
    for (int j = 0; j < 2; ++j) {
      int row0 = m0 + waveM * 32 + i * 16 + sel * 8;
      int col = n0 + waveN * 32 + j * 16 + l16;
      float* cp = C + (size_t)row0 * N + col;
#pragma unroll
      for (int r = 0; r < 8; ++r) {
        *cp = acc[i][j][r];
        cp += N;
      }
    }
}

// ---------------------------------------------------------------------------
// 3) per-head RMSNorm + RoPE. One wave per (t, row) where row in [0,32):
//    0-15 q heads, 16-23 k heads, 24-31 v heads. Lane l owns dims l,l+32,l+64,l+96.
//    V is written TRANSPOSED: vT[h][d][t] for contiguous attention B-fragments.
// ---------------------------------------------------------------------------
__global__ __launch_bounds__(256) void norm_rope_kernel(
    const float* __restrict__ qkv, const int* __restrict__ pos_ids,
    const float* __restrict__ qw, const float* __restrict__ kw,
    unsigned short* __restrict__ qb, unsigned short* __restrict__ kb,
    unsigned short* __restrict__ vT) {
  int gw = blockIdx.x * 8 + (threadIdx.x >> 5);
  int lane = threadIdx.x & 31;
  int t = gw >> 5;
  int r = gw & 31;

  const float* src;
  int kind, h;
  if (r < 16)      { kind = 0; h = r;      src = qkv + (size_t)t * QKV_OUT + h * HEAD_DIM; }
  else if (r < 24) { kind = 1; h = r - 16; src = qkv + (size_t)t * QKV_OUT + 2048 + h * HEAD_DIM; }
  else             { kind = 2; h = r - 24; src = qkv + (size_t)t * QKV_OUT + 3072 + h * HEAD_DIM; }

  float x0 = src[lane], x1 = src[lane + 32], x2 = src[lane + 64], x3 = src[lane + 96];

  if (kind == 2) {  // V: bf16 + transpose
    size_t base = ((size_t)h * HEAD_DIM) * T_SEQ + t;
    vT[base + (size_t)(lane)      * T_SEQ] = f2bf(x0);
    vT[base + (size_t)(lane + 32) * T_SEQ] = f2bf(x1);
    vT[base + (size_t)(lane + 64) * T_SEQ] = f2bf(x2);
    vT[base + (size_t)(lane + 96) * T_SEQ] = f2bf(x3);
    return;
  }

  float ss = x0 * x0 + x1 * x1 + x2 * x2 + x3 * x3;
#pragma unroll
  for (int m = 1; m < 32; m <<= 1) ss += __shfl_xor(ss, m, 32);
  float inv = rsqrtf(ss * (1.0f / 128.0f) + 1e-6f);
  const float* w = (kind == 0) ? qw : kw;
  x0 *= inv * w[lane]; x1 *= inv * w[lane + 32];
  x2 *= inv * w[lane + 64]; x3 *= inv * w[lane + 96];

  // neox RoPE: pairs (i, i+64), inv_freq = base^(-i/64), ln(1e6)=13.8155106
  float p = (float)pos_ids[t];
  float f0 = __expf(-((float)lane)        * (1.0f / 64.0f) * 13.8155106f);
  float f1 = __expf(-((float)(lane + 32)) * (1.0f / 64.0f) * 13.8155106f);
  float a0 = p * f0, a1 = p * f1;
  float c0, s0, c1, s1;
  __sincosf(a0, &s0, &c0);
  __sincosf(a1, &s1, &c1);
  float o0 = x0 * c0 - x2 * s0;
  float o2 = x2 * c0 + x0 * s0;
  float o1 = x1 * c1 - x3 * s1;
  float o3 = x3 * c1 + x1 * s1;

  unsigned short* dst = (kind == 0)
      ? qb + ((size_t)t * NUM_HEADS + h) * HEAD_DIM
      : kb + ((size_t)t * NUM_KV + h) * HEAD_DIM;
  dst[lane] = f2bf(o0); dst[lane + 32] = f2bf(o1);
  dst[lane + 64] = f2bf(o2); dst[lane + 96] = f2bf(o3);
}

// ---------------------------------------------------------------------------
// 4) Causal flash attention. One wave per (head, 16-row q tile); 32 kv rows
//    per step = two 16x16 score tiles so P is a full K=32 A-operand.
// ---------------------------------------------------------------------------
__global__ __launch_bounds__(128) void attn_kernel(
    const unsigned short* __restrict__ qb, const unsigned short* __restrict__ kb,
    const unsigned short* __restrict__ vT, unsigned short* __restrict__ ob) {
  __shared__ unsigned short Ps[4][16][32];  // per-wave P transpose stage
  const int lane = threadIdx.x & 31;
  const int w = threadIdx.x >> 5;
  const int gw = blockIdx.x * 4 + w;
  const int h = gw >> 7;        // 16 heads x 128 q tiles
  const int tq = gw & 127;
  const int m0 = tq * 16;
  const int kvh = h >> 1;       // GQA: 2 q heads per kv head
  const int sel = lane >> 4;
  const int l16 = lane & 15;
  const float scale = 0.08838834764831845f;  // 1/sqrt(128)

  // Q fragments (16 rows x 128 dims = 4 chunks of K=32)
  bf16x16 qf[4];
  {
    const unsigned short* qrow =
        qb + ((size_t)(m0 + l16) * NUM_HEADS + h) * HEAD_DIM;
#pragma unroll
    for (int c = 0; c < 4; ++c) {
      uint4 lo = *(const uint4*)(qrow + c * 32 + sel * 8);
      uint4 hi = *(const uint4*)(qrow + c * 32 + 16 + sel * 8);
      qf[c] = mkfrag(lo, hi);
    }
  }

  f32x8 O[8];
#pragma unroll
  for (int c = 0; c < 8; ++c) O[c] = zero8();
  float mrow[8], lrow[8];
#pragma unroll
  for (int r = 0; r < 8; ++r) { mrow[r] = -1e30f; lrow[r] = 0.f; }

  const int nj = (tq >> 1) + 1;  // 32-wide kv blocks covering rows <= m0+15
  for (int j = 0; j < nj; ++j) {
    const int kv0 = j * 32;

    // S = Q K^T : B-frag lane = kv column, K-contiguous loads from kb rows
    f32x8 S0 = zero8(), S1 = zero8();
    const unsigned short* kr0 =
        kb + ((size_t)(kv0 + l16) * NUM_KV + kvh) * HEAD_DIM;
    const unsigned short* kr1 =
        kb + ((size_t)(kv0 + 16 + l16) * NUM_KV + kvh) * HEAD_DIM;
#pragma unroll
    for (int c = 0; c < 4; ++c) {
      uint4 lo0 = *(const uint4*)(kr0 + c * 32 + sel * 8);
      uint4 hi0 = *(const uint4*)(kr0 + c * 32 + 16 + sel * 8);
      S0 = __builtin_amdgcn_wmma_f32_16x16x32_bf16(
          false, qf[c], false, mkfrag(lo0, hi0), (short)0, S0, false, false);
      uint4 lo1 = *(const uint4*)(kr1 + c * 32 + sel * 8);
      uint4 hi1 = *(const uint4*)(kr1 + c * 32 + 16 + sel * 8);
      S1 = __builtin_amdgcn_wmma_f32_16x16x32_bf16(
          false, qf[c], false, mkfrag(lo1, hi1), (short)0, S1, false, false);
    }

    // scale + causal mask (C layout: lane half sel owns rows sel*8+r, col l16)
#pragma unroll
    for (int r = 0; r < 8; ++r) {
      int row = m0 + sel * 8 + r;
      int c0 = kv0 + l16, c1 = c0 + 16;
      float s0v = S0[r] * scale; if (c0 > row) s0v = -1e30f;
      float s1v = S1[r] * scale; if (c1 > row) s1v = -1e30f;
      S0[r] = s0v; S1[r] = s1v;
    }

    // online softmax; reductions stay within 16-lane halves (xor 1..8)
#pragma unroll
    for (int r = 0; r < 8; ++r) {
      float mx = fmaxf(S0[r], S1[r]);
#pragma unroll
      for (int msk = 1; msk < 16; msk <<= 1) mx = fmaxf(mx, __shfl_xor(mx, msk, 32));
      float mnew = fmaxf(mrow[r], mx);
      float alpha = __expf(mrow[r] - mnew);
      float p0 = __expf(S0[r] - mnew);
      float p1 = __expf(S1[r] - mnew);
      float rs = p0 + p1;
#pragma unroll
      for (int msk = 1; msk < 16; msk <<= 1) rs += __shfl_xor(rs, msk, 32);
      lrow[r] = lrow[r] * alpha + rs;
      mrow[r] = mnew;
      S0[r] = p0; S1[r] = p1;
#pragma unroll
      for (int c = 0; c < 8; ++c) O[c][r] *= alpha;
    }

    // P (C layout) -> A-operand layout via LDS; per-wave LDS is in-order (DScnt)
#pragma unroll
    for (int r = 0; r < 8; ++r) {
      Ps[w][sel * 8 + r][l16]      = f2bf(S0[r]);
      Ps[w][sel * 8 + r][l16 + 16] = f2bf(S1[r]);
    }
    asm volatile("s_wait_dscnt 0x0" ::: "memory");
    bf16x16 pf;
    {
      uint4 lo = *(const uint4*)&Ps[w][l16][sel * 8];
      uint4 hi = *(const uint4*)&Ps[w][l16][16 + sel * 8];
      pf = mkfrag(lo, hi);
    }

    // O += P V : vT[h][d][t] makes each B-fragment lane two contiguous 16B loads (k = 8h.., 16+8h..)
#pragma unroll
    for (int c = 0; c < 8; ++c) {
      const unsigned short* vp =
          vT + ((size_t)(kvh * HEAD_DIM + c * 16 + l16)) * T_SEQ + kv0 + sel * 8;
      uint4 lo = *(const uint4*)(vp);
      uint4 hi = *(const uint4*)(vp + 16);
      O[c] = __builtin_amdgcn_wmma_f32_16x16x32_bf16(
          false, pf, false, mkfrag(lo, hi), (short)0, O[c], false, false);
    }
  }

  // finalize: O / l, write bf16 [T][16*128] for the output projection
#pragma unroll
  for (int r = 0; r < 8; ++r) {
    float invl = 1.0f / lrow[r];
    int row = m0 + sel * 8 + r;
    unsigned short* orow = ob + (size_t)row * HIDDEN + h * HEAD_DIM;
#pragma unroll
    for (int c = 0; c < 8; ++c) orow[c * 16 + l16] = f2bf(O[c][r] * invl);
  }
}

// ---------------------------------------------------------------------------
extern "C" void kernel_launch(void* const* d_in, const int* in_sizes, int n_in,
                              void* d_out, int out_size, void* d_ws,
                              size_t ws_size, hipStream_t stream) {
  const float* x     = (const float*)d_in[0];
  const int*   pos   = (const int*)d_in[1];
  const float* w_qkv = (const float*)d_in[2];
  const float* w_o   = (const float*)d_in[3];
  const float* qw    = (const float*)d_in[4];
  const float* kw    = (const float*)d_in[5];
  float* out = (float*)d_out;

  char* ws = (char*)d_ws;
  unsigned short* xb    = (unsigned short*)(ws);                          //  8 MB
  unsigned short* wqkvb = (unsigned short*)(ws + ((size_t)8  << 20));     // 16 MB
  unsigned short* wob   = (unsigned short*)(ws + ((size_t)24 << 20));     //  8 MB
  float*          qkv   = (float*)         (ws + ((size_t)32 << 20));     // 32 MB
  unsigned short* qb    = (unsigned short*)(ws + ((size_t)64 << 20));     //  8 MB
  unsigned short* kb    = (unsigned short*)(ws + ((size_t)72 << 20));     //  4 MB
  unsigned short* vT    = (unsigned short*)(ws + ((size_t)76 << 20));     //  4 MB
  unsigned short* ob    = (unsigned short*)(ws + ((size_t)80 << 20));     //  8 MB

  int n4;
  n4 = HIDDEN * HIDDEN / 4;
  f32_to_bf16_kernel<<<(n4 + 255) / 256, 256, 0, stream>>>(x, xb, n4);
  n4 = HIDDEN * QKV_OUT / 4;
  f32_to_bf16_kernel<<<(n4 + 255) / 256, 256, 0, stream>>>(w_qkv, wqkvb, n4);
  n4 = HIDDEN * HIDDEN / 4;
  f32_to_bf16_kernel<<<(n4 + 255) / 256, 256, 0, stream>>>(w_o, wob, n4);

  gemm_bf16_kernel<<<dim3(QKV_OUT / 64, T_SEQ / 128), 256, 0, stream>>>(
      xb, wqkvb, qkv, T_SEQ, QKV_OUT, HIDDEN);

  norm_rope_kernel<<<(T_SEQ * 32) / 8, 256, 0, stream>>>(qkv, pos, qw, kw, qb,
                                                         kb, vT);

  attn_kernel<<<(NUM_HEADS * (T_SEQ / 16)) / 4, 128, 0, stream>>>(qb, kb, vT,
                                                                  ob);

  gemm_bf16_kernel<<<dim3(HIDDEN / 64, T_SEQ / 128), 256, 0, stream>>>(
      ob, wob, out, T_SEQ, HIDDEN, HIDDEN);
}
